// RNN_21620865368638
// MI455X (gfx1250) — matrix operand order repaired and checked by reading, not compiled, not run on hardware
//
#include <hip/hip_runtime.h>
#include <hip/hip_bf16.h>

#define T_STEPS 2048
#define BATCH   32
#define HID     512
#define KDIM    512

typedef __attribute__((ext_vector_type(16))) __bf16 v16bf;
typedef __attribute__((ext_vector_type(8)))  __bf16 v8bf;
typedef __attribute__((ext_vector_type(8)))  float  v8f;

union V16 {
    v16bf v;
    v8bf  h[2];
    unsigned short s[16];
};

// float -> bf16 bits, round-to-nearest-even
__device__ __forceinline__ unsigned short f2bf(float f) {
    unsigned u = __float_as_uint(f);
    u += 0x7FFFu + ((u >> 16) & 1u);
    return (unsigned short)(u >> 16);
}

// Async memory->LDS 16-byte copy (CDNA5 GLOBAL_LOAD_ASYNC_TO_LDS_B128,
// tracked by ASYNCcnt). lds_ptr: generic pointer to __shared__; low 32 bits
// of the flat address are the LDS byte offset (ISA 10.2 aperture mapping).
__device__ __forceinline__ void async_ld_b128(void* lds_ptr, const void* gptr) {
    unsigned lds_off = (unsigned)(size_t)lds_ptr;
    asm volatile("global_load_async_to_lds_b128 %0, %1, off"
                 :: "v"(lds_off), "v"(gptr) : "memory");
}
__device__ __forceinline__ void wait_async0() {
    asm volatile("s_wait_asynccnt 0x0" ::: "memory");
}

// ---------------------------------------------------------------------------
// Bulk fp32 -> bf16 conversion (vectorized, one-shot per buffer).
// ---------------------------------------------------------------------------
__global__ __launch_bounds__(256) void to_bf16(
    const float* __restrict__ src, unsigned short* __restrict__ dst, int n4) {
    int i = blockIdx.x * blockDim.x + threadIdx.x;
    if (i >= n4) return;
    float4 f = reinterpret_cast<const float4*>(src)[i];
    ushort4 o;
    o.x = f2bf(f.x); o.y = f2bf(f.y); o.z = f2bf(f.z); o.w = f2bf(f.w);
    reinterpret_cast<ushort4*>(dst)[i] = o;
}

// ---------------------------------------------------------------------------
// Zero the (bf16) double-buffered h state and the grid-barrier counter.
// ---------------------------------------------------------------------------
__global__ void init_state(unsigned* __restrict__ h32, unsigned* __restrict__ ctr) {
    int i = blockIdx.x * blockDim.x + threadIdx.x;
    if (i < (2 * BATCH * HID) / 2) h32[i] = 0u;   // bf16 zero bits
    if (i == 0) *ctr = 0u;
}

// ---------------------------------------------------------------------------
// Input projection: Y[m,n] = sum_k X[m,k] * W[n,k] + (b0[n] + b1[n])
// X, W pre-converted bf16. One block = 16 output rows (async-staged to LDS);
// each of 8 waves owns a 16x32 tile (two WMMAs per k-step sharing one A).
// ---------------------------------------------------------------------------
__global__ __launch_bounds__(256) void gemm_inproj(
    const unsigned short* __restrict__ Xb,   // [M, KDIM] bf16
    const unsigned short* __restrict__ Wb,   // [HID, KDIM] bf16 (out x in)
    const float* __restrict__ b0,
    const float* __restrict__ b1,
    float* __restrict__ Y)                   // [M, HID] fp32
{
    __shared__ __attribute__((aligned(16))) unsigned short sA[16][KDIM]; // 16 KB

    const int m0 = blockIdx.x * 16;

    // Stage 16 bf16 rows of X into LDS with async B128 copies.
    {
        const uint4* src = reinterpret_cast<const uint4*>(Xb + (size_t)m0 * KDIM);
        uint4* dst = reinterpret_cast<uint4*>(&sA[0][0]);
        for (int i = threadIdx.x; i < (16 * KDIM) / 8; i += 256)
            async_ld_b128(&dst[i], &src[i]);
        wait_async0();
    }
    __syncthreads();

    const int lane = threadIdx.x & 31;
    const int wave = threadIdx.x >> 5;
    const int hi   = lane >> 4;   // lane half (0: lanes 0-15, 1: lanes 16-31)
    const int r16  = lane & 15;

    for (int nbase = 0; nbase < HID; nbase += 256) {
        const int n0 = nbase + wave * 32;            // 32 output cols per wave
        const unsigned short* wp0 = Wb + (size_t)(n0 + r16) * KDIM;
        const unsigned short* wp1 = Wb + (size_t)(n0 + 16 + r16) * KDIM;
        __builtin_prefetch(wp0, 0, 1);
        __builtin_prefetch(wp1, 0, 1);

        v8f c0 = {}, c1 = {};
        #pragma unroll
        for (int ks = 0; ks < KDIM / 32; ++ks) {
            V16 a, bA, bB;
            // A: row M = r16; per-lane K = hi*8 + {0..7} and +16.
            a.h[0] = *reinterpret_cast<const v8bf*>(&sA[r16][ks * 32 + hi * 8]);
            a.h[1] = *reinterpret_cast<const v8bf*>(&sA[r16][ks * 32 + hi * 8 + 16]);
            // B: col N = r16; per-lane K = hi*8 + {0..7} and +16.
            bA.h[0] = *reinterpret_cast<const v8bf*>(wp0 + ks * 32 + hi * 8);
            bA.h[1] = *reinterpret_cast<const v8bf*>(wp0 + ks * 32 + hi * 8 + 16);
            bB.h[0] = *reinterpret_cast<const v8bf*>(wp1 + ks * 32 + hi * 8);
            bB.h[1] = *reinterpret_cast<const v8bf*>(wp1 + ks * 32 + hi * 8 + 16);
            c0 = __builtin_amdgcn_wmma_f32_16x16x32_bf16(
                     false, a.v, false, bA.v, (short)0, c0, false, false);
            c1 = __builtin_amdgcn_wmma_f32_16x16x32_bf16(
                     false, a.v, false, bB.v, (short)0, c1, false, false);
        }

        const float bias0 = b0[n0 + r16]      + b1[n0 + r16];
        const float bias1 = b0[n0 + 16 + r16] + b1[n0 + 16 + r16];
        #pragma unroll
        for (int r = 0; r < 8; ++r) {
            int m = m0 + r + hi * 8;               // C/D layout: M = r + 8*hi
            Y[(size_t)m * HID + n0 + r16]      = c0[r] + bias0;
            Y[(size_t)m * HID + n0 + 16 + r16] = c1[r] + bias1;
        }
    }
}

// ---------------------------------------------------------------------------
// Persistent recurrent scan: 16 blocks, each owns 32 output columns and keeps
// its bf16 W_hh slice packed in LDS in B-operand order for the whole scan.
// h state lives in global as bf16 (ping-pong), async-staged into LDS each
// step; per-step device-wide barrier via atomics.
// ---------------------------------------------------------------------------
__global__ __launch_bounds__(128) void rnn_scan(
    const float* __restrict__ XW,            // [T, B, H] input projection
    const float* __restrict__ Whh,           // [H, H] fp32
    unsigned short* __restrict__ Hst,        // [2, B, H] bf16 state (zeroed)
    float* __restrict__ Out,                 // [T, B, H] fp32
    unsigned* __restrict__ ctr)              // grid barrier counter (zeroed)
{
    __shared__ __attribute__((aligned(32))) unsigned short sW[2][16][32][16]; // 32 KB
    __shared__ __attribute__((aligned(16))) unsigned short sH[BATCH][HID];    // 32 KB

    const int n0    = blockIdx.x * 32;
    const int lane  = threadIdx.x & 31;
    const int wave  = threadIdx.x >> 5;
    const int mtile = wave & 1;     // batch rows 0-15 / 16-31
    const int ntile = wave >> 1;    // cols n0+0-15 / n0+16-31
    const int hi    = lane >> 4;
    const int r16   = lane & 15;

    // One-time pack of this block's W_hh slice into B-operand order:
    // sW[nt][ks][lane][j] = Whh[n0 + nt*16 + (lane&15)][ks*32 + (lane>>4)*16 + j]
    for (int i = threadIdx.x; i < 2 * 16 * 32 * 16; i += 128) {
        int j  = i & 15;
        int ln = (i >> 4) & 31;
        int ks = (i >> 9) & 15;
        int nt = i >> 13;
        int n  = n0 + nt * 16 + (ln & 15);
        int k  = ks * 32 + (ln >> 4) * 16 + j;
        sW[nt][ks][ln][j] = f2bf(Whh[(size_t)n * HID + k]);
    }

    const unsigned nblocks = gridDim.x;

    for (int t = 0; t < T_STEPS; ++t) {
        __syncthreads();
        // Async-stage previous bf16 h (32x512 = 32 KB) into LDS.
        {
            const uint4* src = reinterpret_cast<const uint4*>(
                Hst + (size_t)(t & 1) * BATCH * HID);
            uint4* dst = reinterpret_cast<uint4*>(&sH[0][0]);
            for (int i = threadIdx.x; i < (BATCH * HID) / 8; i += 128)
                async_ld_b128(&dst[i], &src[i]);
            wait_async0();
        }
        __syncthreads();

        // C initialized with the xw_t tile (fp32 accumulator).
        const float* xw = XW + (size_t)t * BATCH * HID;
        v8f c;
        #pragma unroll
        for (int r = 0; r < 8; ++r) {
            int m = mtile * 16 + r + hi * 8;
            c[r] = xw[(size_t)m * HID + n0 + ntile * 16 + r16];
        }

        #pragma unroll
        for (int ks = 0; ks < HID / 32; ++ks) {
            V16 a, b;
            a.h[0] = *reinterpret_cast<const v8bf*>(&sH[mtile * 16 + r16][ks * 32 + hi * 16]);
            a.h[1] = *reinterpret_cast<const v8bf*>(&sH[mtile * 16 + r16][ks * 32 + hi * 16 + 8]);
            b.v    = *reinterpret_cast<const v16bf*>(&sW[ntile][ks][lane][0]);
            c = __builtin_amdgcn_wmma_f32_16x16x32_bf16(
                    false, a.v, false, b.v, (short)0, c, false, false);
        }

        // tanh; fp32 to Out, bf16 to next-step state buffer.
        unsigned short* hout = Hst + (size_t)((t + 1) & 1) * BATCH * HID;
        float* outp = Out + (size_t)t * BATCH * HID;
        #pragma unroll
        for (int r = 0; r < 8; ++r) {
            int m = mtile * 16 + r + hi * 8;
            float v = tanhf(c[r]);
            size_t idx = (size_t)m * HID + n0 + ntile * 16 + r16;
            outp[idx] = v;
            hout[idx] = f2bf(v);
        }

        // Device-wide step barrier: make writes visible, arrive, spin.
        __threadfence();
        __syncthreads();
        if (threadIdx.x == 0) {
            atomicAdd(ctr, 1u);
            const unsigned target = nblocks * (unsigned)(t + 1);
            while (__hip_atomic_load(ctr, __ATOMIC_RELAXED,
                                     __HIP_MEMORY_SCOPE_AGENT) < target) {
                __builtin_amdgcn_s_sleep(1);
            }
        }
        __syncthreads();
    }
}

// ---------------------------------------------------------------------------
extern "C" void kernel_launch(void* const* d_in, const int* in_sizes, int n_in,
                              void* d_out, int out_size, void* d_ws, size_t ws_size,
                              hipStream_t stream) {
    (void)in_sizes; (void)n_in; (void)out_size; (void)ws_size;

    const float* x     = (const float*)d_in[0];
    const float* w_ih0 = (const float*)d_in[1];
    const float* w_hh0 = (const float*)d_in[2];
    const float* b_ih0 = (const float*)d_in[3];
    const float* b_hh0 = (const float*)d_in[4];
    const float* w_ih1 = (const float*)d_in[5];
    const float* w_hh1 = (const float*)d_in[6];
    const float* b_ih1 = (const float*)d_in[7];
    const float* b_hh1 = (const float*)d_in[8];
    float* out = (float*)d_out;

    const size_t TBH = (size_t)T_STEPS * BATCH * HID;       // 33,554,432
    char* ws = (char*)d_ws;
    float*          XW   = (float*)ws;                                   // 128 MB
    float*          OUT0 = (float*)(ws + TBH * 4);                       // 128 MB
    unsigned short* XB   = (unsigned short*)(ws + 2 * TBH * 4);          //  64 MB
    unsigned short* WB   = (unsigned short*)(ws + 2 * TBH * 4 + TBH * 2);// 512 KB
    unsigned short* HST  = (unsigned short*)(ws + 2 * TBH * 4 + TBH * 2
                                                + (size_t)HID * KDIM * 2);
    unsigned*       ctr  = (unsigned*)(ws + 2 * TBH * 4 + TBH * 2
                                          + (size_t)HID * KDIM * 2
                                          + (size_t)2 * BATCH * HID * 2);

    const int Mtiles = (T_STEPS * BATCH) / 16;              // 4096 blocks
    const int cvtX   = (int)(TBH / 4) / 256;                // 32768 blocks
    const int cvtW   = (HID * KDIM / 4) / 256;              // 256 blocks

    // Layer 0
    to_bf16<<<cvtX, 256, 0, stream>>>(x, XB, (int)(TBH / 4));
    to_bf16<<<cvtW, 256, 0, stream>>>(w_ih0, WB, HID * KDIM / 4);
    init_state<<<128, 256, 0, stream>>>((unsigned*)HST, ctr);
    gemm_inproj<<<Mtiles, 256, 0, stream>>>(XB, WB, b_ih0, b_hh0, XW);
    rnn_scan<<<16, 128, 0, stream>>>(XW, w_hh0, HST, OUT0, ctr);

    // Layer 1
    to_bf16<<<cvtX, 256, 0, stream>>>(OUT0, XB, (int)(TBH / 4));
    to_bf16<<<cvtW, 256, 0, stream>>>(w_ih1, WB, HID * KDIM / 4);
    init_state<<<128, 256, 0, stream>>>((unsigned*)HST, ctr);
    gemm_inproj<<<Mtiles, 256, 0, stream>>>(XB, WB, b_ih1, b_hh1, XW);
    rnn_scan<<<16, 128, 0, stream>>>(XW, w_hh1, HST, out, ctr);
}
